// NMSWrapper_30571577213232
// MI455X (gfx1250) — 8-bit matrix operand order repaired and checked by reading, compiled, not run on hardware
//
#include <hip/hip_runtime.h>
#include <stdint.h>

#define B_     16
#define N_     25200
#define C_     80
#define NC_    (N_ * C_)
#define KPT_   51
#define KPRE   1024
#define MAXDET 300
#define CAP    8192
#define CCAP   65536
#define NB1    4096
#define NMSCH  512
#define IOU_THR 0.7f
#define SCORE_THR 0.001f

typedef int   v16i __attribute__((ext_vector_type(16)));
typedef float v8f  __attribute__((ext_vector_type(8)));

__global__ void k_zero_u32(uint32_t* p, int n) {
  int g = blockIdx.x * blockDim.x + threadIdx.x;
  if (g < n) p[g] = 0u;
}
__global__ void k_zero_f32(float* p, int n) {
  int g = blockIdx.x * blockDim.x + threadIdx.x;
  if (g < n) p[g] = 0.0f;
}

__global__ __launch_bounds__(256) void k_hist(const float4* __restrict__ scores4,
                                              uint32_t* __restrict__ hist) {
  int b = blockIdx.y;
  __shared__ uint32_t lh[NB1];
  for (int i = threadIdx.x; i < NB1; i += 256) lh[i] = 0u;
  __syncthreads();
  const float4* p = scores4 + (size_t)b * (NC_ / 4);
  int stride = gridDim.x * 256;
  for (int i = blockIdx.x * 256 + threadIdx.x; i < NC_ / 4; i += stride) {
    float4 v = p[i];
    atomicAdd(&lh[__float_as_uint(v.x) >> 20], 1u);
    atomicAdd(&lh[__float_as_uint(v.y) >> 20], 1u);
    atomicAdd(&lh[__float_as_uint(v.z) >> 20], 1u);
    atomicAdd(&lh[__float_as_uint(v.w) >> 20], 1u);
  }
  __syncthreads();
  uint32_t* gh = hist + b * NB1;
  for (int i = threadIdx.x; i < NB1; i += 256) {
    uint32_t c = lh[i];
    if (c) atomicAdd(&gh[i], c);
  }
}

__global__ __launch_bounds__(256) void k_thresh(const uint32_t* __restrict__ hist,
                                                uint32_t* __restrict__ l1bin) {
  int b = blockIdx.x, t = threadIdx.x;
  __shared__ uint32_t part[256];
  __shared__ uint32_t pref[256];
  const uint32_t* h = hist + b * NB1;
  uint32_t s = 0;
  for (int r = 0; r < 16; ++r) s += h[NB1 - 1 - (t * 16 + r)];
  part[t] = s;
  __syncthreads();
  if (t == 0) {
    uint32_t a = 0;
    for (int i = 0; i < 256; ++i) { pref[i] = a; a += part[i]; }
  }
  __syncthreads();
  if (pref[t] < KPRE && pref[t] + part[t] >= KPRE) {
    uint32_t acc = pref[t];
    for (int r = 0; r < 16; ++r) {
      uint32_t bin = NB1 - 1 - (t * 16 + r);
      acc += h[bin];
      if (acc >= KPRE) { l1bin[b] = bin; break; }
    }
  }
}

__global__ __launch_bounds__(256) void k_ccollect(const float4* __restrict__ scores4,
                                                  const uint32_t* __restrict__ l1bin,
                                                  uint32_t* __restrict__ ccnt,
                                                  uint64_t* __restrict__ coarse) {
  int b = blockIdx.y;
  uint32_t thr = l1bin[b];
  const float4* p = scores4 + (size_t)b * (NC_ / 4);
  int stride = gridDim.x * 256;
  for (int i = blockIdx.x * 256 + threadIdx.x; i < NC_ / 4; i += stride) {
    float4 v = p[i];
    uint32_t bb[4] = { __float_as_uint(v.x), __float_as_uint(v.y),
                       __float_as_uint(v.z), __float_as_uint(v.w) };
#pragma unroll
    for (int q = 0; q < 4; ++q) {
      if ((bb[q] >> 20) >= thr) {
        uint32_t pos = atomicAdd(&ccnt[b], 1u);
        if (pos < CCAP)
          coarse[(size_t)b * CCAP + pos] =
              ((uint64_t)bb[q] << 32) |
              (uint32_t)(0xFFFFFFFFu - (uint32_t)(i * 4 + q));
      }
    }
  }
}

__global__ __launch_bounds__(1024) void k_select(const uint64_t* __restrict__ coarse,
                                                 const uint32_t* __restrict__ ccnt,
                                                 const uint32_t* __restrict__ l1bin,
                                                 const float* __restrict__ boxes,
                                                 float* __restrict__ top_s,
                                                 int* __restrict__ top_cls,
                                                 int* __restrict__ top_boxi,
                                                 float* __restrict__ cbout) {
  int b = blockIdx.x, tid = threadIdx.x;
  __shared__ uint64_t key[CAP];
  __shared__ uint32_t s_nhi, s_cut, s_cnt;
  uint32_t* fh = (uint32_t*)key;
  for (int i = tid; i < NB1; i += 1024) fh[i] = 0u;
  if (tid == 0) { s_nhi = 0u; s_cut = 0u; s_cnt = 0u; }
  __syncthreads();

  int n = (int)min(ccnt[b], (uint32_t)CCAP);
  uint32_t l1 = l1bin[b];
  const uint64_t* cp = coarse + (size_t)b * CCAP;

  for (int i = tid; i < n; i += 1024) {
    uint32_t bits = (uint32_t)(cp[i] >> 32);
    if ((bits >> 20) > l1) atomicAdd(&s_nhi, 1u);
    else                   atomicAdd(&fh[(bits >> 8) & 0xFFFu], 1u);
  }
  __syncthreads();
  if (tid == 0) {
    if (s_nhi >= KPRE) s_cut = NB1;
    else {
      uint32_t need = KPRE - s_nhi, acc = 0, cut = 0;
      for (int f = NB1 - 1; f >= 0; --f) {
        acc += fh[f];
        if (acc >= need) { cut = (uint32_t)f; break; }
      }
      s_cut = cut;
    }
  }
  __syncthreads();
  uint32_t cut = s_cut;
  __syncthreads();
  for (int i = tid; i < CAP; i += 1024) key[i] = 0ull;
  __syncthreads();
  for (int i = tid; i < n; i += 1024) {
    uint64_t kv = cp[i];
    uint32_t bits = (uint32_t)(kv >> 32);
    bool take = ((bits >> 20) > l1) || (((bits >> 8) & 0xFFFu) >= cut);
    if (take) {
      uint32_t pos = atomicAdd(&s_cnt, 1u);
      if (pos < CAP) key[pos] = kv;
    }
  }
  __syncthreads();

  for (int k = 2; k <= CAP; k <<= 1) {
    for (int j = k >> 1; j > 0; j >>= 1) {
      for (int p = tid; p < CAP; p += 1024) {
        int ixj = p ^ j;
        if (ixj > p) {
          uint64_t a = key[p], c = key[ixj];
          bool desc = ((p & k) == 0);
          if (desc ? (a < c) : (a > c)) { key[p] = c; key[ixj] = a; }
        }
      }
      __syncthreads();
    }
  }

  if (tid < KPRE) {
    uint64_t kv = key[tid];
    uint32_t bits = (uint32_t)(kv >> 32);
    uint32_t idx  = 0xFFFFFFFFu - (uint32_t)(kv & 0xFFFFFFFFu);
    if (kv == 0ull) { bits = 0u; idx = 0u; }
    int bi = (int)(idx / C_);
    int cl = (int)(idx % C_);
    if (bi >= N_) bi = 0;
    float4 bx = ((const float4*)boxes)[b * N_ + bi];
    top_s[b * KPRE + tid]    = __uint_as_float(bits);
    top_cls[b * KPRE + tid]  = cl;
    top_boxi[b * KPRE + tid] = bi;
    ((float4*)cbout)[b * KPRE + tid] = bx;
  }
}

__global__ __launch_bounds__(32) void k_sup(const float* __restrict__ cb,
                                            const int* __restrict__ cls,
                                            uint32_t* __restrict__ sup) {
  int b = blockIdx.x >> 6;
  int it = blockIdx.x & 63;
  int i_base = it * 16;
  int lane = threadIdx.x;
  int m = lane & 15;
  int h = lane >> 4;
  __shared__ float4  ibox[16];
  __shared__ float   iarea[16];
  __shared__ int     icls[16];
  __shared__ uint32_t supw[16][32];

  if (lane < 16) {
    float4 bx = ((const float4*)cb)[b * KPRE + i_base + lane];
    ibox[lane]  = bx;
    iarea[lane] = fmaxf(bx.z - bx.x, 0.f) * fmaxf(bx.w - bx.y, 0.f);
    icls[lane]  = cls[b * KPRE + i_base + lane];
  }
  for (int w = lane; w < 16 * 32; w += 32) ((uint32_t*)supw)[w] = 0u;
  __syncthreads();

  int ci = icls[m];
  v16i A;
#pragma unroll
  for (int v = 0; v < 16; ++v) {
    int vv = v & 7;
    int kbase = ((v >> 3) << 6) + ((vv >> 1) << 4) + ((vv & 1) << 2) + (h << 3);
    int d = ci - kbase;
    A[v] = (d >= 0 && d < 4) ? (0x38 << (d * 8)) : 0;
  }

  for (int jt = 0; jt < 64; ++jt) {
    int j_base = jt * 16;
    int jg = j_base + m;
    int cj = cls[b * KPRE + jg];
    float4 jb = ((const float4*)cb)[b * KPRE + jg];
    float jarea = fmaxf(jb.z - jb.x, 0.f) * fmaxf(jb.w - jb.y, 0.f);
    v16i Bf;
#pragma unroll
    for (int v = 0; v < 16; ++v) {
      int kbase = ((v >> 3) << 6) + (((v & 7) >> 1) << 4) + ((v & 1) << 2) + (h << 3);
      int d = cj - kbase;
      Bf[v] = (d >= 0 && d < 4) ? (0x38 << (d * 8)) : 0;
    }
    v8f Cz = {};
    v8f D = __builtin_amdgcn_wmma_f32_16x16x128_fp8_fp8(A, Bf, (short)0, Cz, false, false);

#pragma unroll
    for (int r = 0; r < 8; ++r) {
      int M = r + (h << 3);
      float4 ib = ibox[M];
      float lx = fmaxf(ib.x, jb.x), ly = fmaxf(ib.y, jb.y);
      float rx = fminf(ib.z, jb.z), ry = fminf(ib.w, jb.w);
      float inter = fmaxf(rx - lx, 0.f) * fmaxf(ry - ly, 0.f);
      float uni = iarea[M] + jarea - inter;
      bool pred = (inter > IOU_THR * fmaxf(uni, 1e-9f)) &&
                  (D[r] > 0.5f) && (jg > i_base + M);
      uint32_t msk = __builtin_amdgcn_ballot_w32(pred);
      if (lane == 0) {
        int wi = j_base >> 5, sh = j_base & 31;
        supw[r][wi]     |= (msk & 0xFFFFu) << sh;
        supw[r + 8][wi] |= (msk >> 16) << sh;
      }
    }
  }
  __syncthreads();
  for (int w = lane; w < 16 * 32; w += 32) {
    int row = w >> 5, col = w & 31;
    sup[(b * KPRE + i_base + row) * 32 + col] = supw[row][col];
  }
}

__global__ __launch_bounds__(1024) void k_nms(const float* __restrict__ top_s,
                                              const uint32_t* __restrict__ sup,
                                              uint32_t* __restrict__ keep,
                                              int chunk) {
  int b = blockIdx.x, tid = threadIdx.x;
  int lane = tid & 31, wv = tid >> 5;
  __shared__ uint32_t lsup[NMSCH * 32];
  const uint4* src = (const uint4*)(sup + ((size_t)b * KPRE + chunk * NMSCH) * 32);
  uint4* dst = (uint4*)lsup;
  for (int i = tid; i < NMSCH * 32 / 4; i += 1024) dst[i] = src[i];
  __syncthreads();
  if (wv == 0) {
    uint32_t w;
    if (chunk == 0) {
      w = 0u;
      for (int t = 0; t < 32; ++t)
        if (top_s[b * KPRE + lane * 32 + t] > SCORE_THR) w |= (1u << t);
    } else {
      w = keep[b * 32 + lane];
    }
    for (int ii = 0; ii < NMSCH; ++ii) {
      int i = chunk * NMSCH + ii;
      uint32_t cw = __shfl(w, i >> 5, 32);
      if ((cw >> (i & 31)) & 1u)
        w &= ~lsup[ii * 32 + lane];
    }
    keep[b * 32 + lane] = w;
  }
}

__global__ __launch_bounds__(1024) void k_final(const float* __restrict__ top_s,
                                                const int* __restrict__ top_cls,
                                                const int* __restrict__ top_boxi,
                                                const float* __restrict__ cb,
                                                const uint32_t* __restrict__ keep,
                                                const float* __restrict__ kpts,
                                                float* __restrict__ out) {
  int b = blockIdx.x, tid = threadIdx.x;
  __shared__ int pf[1024];
  int flag = (keep[b * 32 + (tid >> 5)] >> (tid & 31)) & 1;
  pf[tid] = flag;
  __syncthreads();
  for (int off = 1; off < 1024; off <<= 1) {
    int v = (tid >= off) ? pf[tid - off] : 0;
    __syncthreads();
    pf[tid] += v;
    __syncthreads();
  }
  int pos = pf[tid] - flag;
  if (flag && pos < MAXDET) {
    float* ob = out;
    float* os = out + B_ * MAXDET * 4;
    int*   ol = (int*)(out + B_ * MAXDET * 5);
    float* ok = out + B_ * MAXDET * 6;
    float4 bx = ((const float4*)cb)[b * KPRE + tid];
    ob[(b * MAXDET + pos) * 4 + 0] = bx.x;
    ob[(b * MAXDET + pos) * 4 + 1] = bx.y;
    ob[(b * MAXDET + pos) * 4 + 2] = bx.z;
    ob[(b * MAXDET + pos) * 4 + 3] = bx.w;
    os[b * MAXDET + pos] = top_s[b * KPRE + tid];
    ol[b * MAXDET + pos] = top_cls[b * KPRE + tid];
    int bi = top_boxi[b * KPRE + tid];
    const float* kp = kpts + ((long)b * N_ + bi) * KPT_;
    float* okp = ok + (long)(b * MAXDET + pos) * KPT_;
    for (int t = 0; t < KPT_; ++t) okp[t] = kp[t];
  }
}

extern "C" void kernel_launch(void* const* d_in, const int* in_sizes, int n_in,
                              void* d_out, int out_size, void* d_ws, size_t ws_size,
                              hipStream_t stream) {
  (void)in_sizes; (void)n_in; (void)ws_size;
  const float* boxes  = (const float*)d_in[0];
  const float* scores = (const float*)d_in[1];
  const float* kpts   = (const float*)d_in[2];
  float* out = (float*)d_out;

  char* ws = (char*)d_ws;
  uint32_t* hist     = (uint32_t*)(ws + 0);
  uint32_t* ccnt     = (uint32_t*)(ws + 262144);
  uint32_t* l1bin    = (uint32_t*)(ws + 262208);
  uint64_t* coarse   = (uint64_t*)(ws + 262272);
  float*    top_s    = (float*)   (ws + 8650880);
  int*      top_cls  = (int*)     (ws + 8716416);
  int*      top_boxi = (int*)     (ws + 8781952);
  float*    cb       = (float*)   (ws + 8847488);
  uint32_t* sup      = (uint32_t*)(ws + 9109632);
  uint32_t* keep     = (uint32_t*)(ws + 11206784);

  {
    int nw = B_ * NB1 + 2 * B_;
    k_zero_u32<<<(nw + 255) / 256, 256, 0, stream>>>(hist, nw);
    k_zero_f32<<<(out_size + 255) / 256, 256, 0, stream>>>(out, out_size);
  }
  k_hist<<<dim3(128, B_), 256, 0, stream>>>((const float4*)scores, hist);
  k_thresh<<<B_, 256, 0, stream>>>(hist, l1bin);
  k_ccollect<<<dim3(128, B_), 256, 0, stream>>>((const float4*)scores, l1bin, ccnt, coarse);
  k_select<<<B_, 1024, 0, stream>>>(coarse, ccnt, l1bin, boxes,
                                    top_s, top_cls, top_boxi, cb);
  k_sup<<<B_ * 64, 32, 0, stream>>>(cb, top_cls, sup);
  k_nms<<<B_, 1024, 0, stream>>>(top_s, sup, keep, 0);
  k_nms<<<B_, 1024, 0, stream>>>(top_s, sup, keep, 1);
  k_final<<<B_, 1024, 0, stream>>>(top_s, top_cls, top_boxi, cb, keep, kpts, out);
}
